// DDNerfModel_57526791962952
// MI455X (gfx1250) — matrix operand order repaired and checked by reading, not compiled, not run on hardware
//
#include <hip/hip_runtime.h>
#include <hip/hip_bf16.h>

// ---------------------------------------------------------------------------
// DD-NeRF forward on gfx1250: f16 WMMA GEMMs for the MLPs, VALU for the rest.
// ---------------------------------------------------------------------------

typedef __attribute__((ext_vector_type(16))) _Float16 v16h;
typedef __attribute__((ext_vector_type(8)))  float    v8f;

#define NRAYS   2048
#define NS      128              // samples per ray (coarse and fine)
#define ENC_DIM 96               // 16 degrees * 3 dims * (sin,cos)
#define HID     256
#define HDIM    128              // width after Wd
#define NEAR_T  2.0f
#define FAR_T   6.0f
#define CH_RAYS 512              // rays per MLP chunk
#define MC      (CH_RAYS * NS)   // rows per chunk = 65536

union FragH { v16h h; float4 f4[2]; };

static __device__ __forceinline__ float sigmoidf_(float x) {
    return 1.0f / (1.0f + expf(-x));
}
static __device__ __forceinline__ float acdf_(float x) {
    return 0.5f * (1.0f + tanhf(0.7978845608028654f * (x + 0.044715f * x * x * x)));
}
static __device__ float erfinv_(float x) {
    // Giles (2010) single-precision approximation
    float w = -logf((1.0f - x) * (1.0f + x));
    float p;
    if (w < 5.0f) {
        w -= 2.5f;
        p = 2.81022636e-08f;
        p = 3.43273939e-07f + p * w;
        p = -3.5233877e-06f + p * w;
        p = -4.39150654e-06f + p * w;
        p = 0.00021858087f + p * w;
        p = -0.00125372503f + p * w;
        p = -0.00417768164f + p * w;
        p = 0.246640727f + p * w;
        p = 1.50140941f + p * w;
    } else {
        w = sqrtf(w) - 3.0f;
        p = -0.000200214257f;
        p = 0.000100950558f + p * w;
        p = 0.00134934322f + p * w;
        p = -0.00367342844f + p * w;
        p = 0.00573950773f + p * w;
        p = -0.0076224613f + p * w;
        p = 0.00943887047f + p * w;
        p = 1.00167406f + p * w;
        p = 2.83297682f + p * w;
    }
    return p * x;
}

// ---------------------------------------------------------------------------
// f32 -> f16 weight conversion
// ---------------------------------------------------------------------------
__global__ void cvt_kernel(const float* __restrict__ src, _Float16* __restrict__ dst, int n) {
    int i = blockIdx.x * blockDim.x + threadIdx.x;
    if (i < n) dst[i] = (_Float16)src[i];
}

// ---------------------------------------------------------------------------
// Per-sample preprocessing: cast_rays + integrated positional encoding.
// One thread per (ray, sample) in the current chunk; writes f16 enc[MC, 96].
// ---------------------------------------------------------------------------
__global__ void __launch_bounds__(256) preproc_kernel(
    const float* __restrict__ ro, const float* __restrict__ rd,
    const float* __restrict__ rr, const float* __restrict__ tfine,
    _Float16* __restrict__ enc, int rowbase)
{
    int id = blockIdx.x * 256 + threadIdx.x;           // local row in chunk
    int grow = rowbase + id;                            // global row
    int r = grow >> 7;                                  // ray
    int s = grow & (NS - 1);                            // sample
    float t0, t1;
    if (tfine) {
        t0 = tfine[r * (NS + 1) + s];
        t1 = tfine[r * (NS + 1) + s + 1];
    } else {
        t0 = NEAR_T + (FAR_T - NEAR_T) * (float)s / (float)NS;
        t1 = NEAR_T + (FAR_T - NEAR_T) * (float)(s + 1) / (float)NS;
    }
    float c  = 0.5f * (t0 + t1);
    float dd = 0.5f * (t1 - t0);
    float d2s = dd * dd, d4 = d2s * d2s, c2 = c * c;
    float denom = 3.0f * c2 + d2s;
    float t_mean = c + 2.0f * c * d2s / denom;
    float t_var  = d2s * (1.0f / 3.0f)
                 - (4.0f / 15.0f) * (d4 * (12.0f * c2 - d2s)) / (denom * denom);
    float rrv = rr[r];
    float r_var = rrv * rrv * (c2 * 0.25f + (5.0f / 12.0f) * d2s
                 - (4.0f / 15.0f) * d4 / denom);

    float rdv0 = rd[r * 3], rdv1 = rd[r * 3 + 1], rdv2 = rd[r * 3 + 2];
    float rov0 = ro[r * 3], rov1 = ro[r * 3 + 1], rov2 = ro[r * 3 + 2];
    float d2 = rdv0 * rdv0 + rdv1 * rdv1 + rdv2 * rdv2;
    float invd2 = 1.0f / fmaxf(d2, 1e-10f);

    float mean[3], cov[3];
    float rdv[3] = {rdv0, rdv1, rdv2};
    float rov[3] = {rov0, rov1, rov2};
    #pragma unroll
    for (int k = 0; k < 3; ++k) {
        mean[k] = rov[k] + rdv[k] * t_mean;
        float douter = rdv[k] * rdv[k];
        cov[k] = t_var * douter + r_var * (1.0f - douter * invd2);
    }

    __align__(16) _Float16 buf[ENC_DIM];
    float scale = 1.0f;
    #pragma unroll
    for (int j = 0; j < 16; ++j) {
        float s2 = scale * scale;
        #pragma unroll
        for (int k = 0; k < 3; ++k) {
            float y   = mean[k] * scale;
            float att = expf(-0.5f * cov[k] * s2);
            buf[j * 3 + k]      = (_Float16)(sinf(y) * att);
            buf[48 + j * 3 + k] = (_Float16)(cosf(y) * att);
        }
        scale *= 2.0f;
    }
    float4* dst = (float4*)(enc + (size_t)id * ENC_DIM);
    const float4* srcb = (const float4*)buf;
    #pragma unroll
    for (int i = 0; i < 12; ++i) dst[i] = srcb[i];
}

// ---------------------------------------------------------------------------
// Per-ray view-direction embedding: direnc[N, 27]
// ---------------------------------------------------------------------------
__global__ void direnc_kernel(const float* __restrict__ rd, float* __restrict__ direnc) {
    int r = blockIdx.x * blockDim.x + threadIdx.x;
    if (r >= NRAYS) return;
    float d0 = rd[r * 3], d1 = rd[r * 3 + 1], d2 = rd[r * 3 + 2];
    float inv = rsqrtf(d0 * d0 + d1 * d1 + d2 * d2);
    float v[3] = {d0 * inv, d1 * inv, d2 * inv};
    float* o = direnc + r * 27;
    o[0] = v[0]; o[1] = v[1]; o[2] = v[2];
    float f = 1.0f;
    #pragma unroll
    for (int j = 0; j < 4; ++j) {
        #pragma unroll
        for (int k = 0; k < 3; ++k) {
            float x = v[k] * f;
            o[3 + j * 3 + k]  = sinf(x);
            o[15 + j * 3 + k] = cosf(x);
        }
        f *= 2.0f;
    }
}

// ---------------------------------------------------------------------------
// Per-ray partial for the concat layer: P[rlocal,128] = direnc @ Wd[256:283] + bd
// ---------------------------------------------------------------------------
__global__ void pbias_kernel(const float* __restrict__ direnc,
                             const float* __restrict__ Wd /*283x128 row-major*/,
                             const float* __restrict__ bd,
                             float* __restrict__ P, int ray0)
{
    int id = blockIdx.x * blockDim.x + threadIdx.x;  // CH_RAYS*128
    int rl = id >> 7, n = id & 127;
    int r = ray0 + rl;
    float s = bd[n];
    #pragma unroll
    for (int j = 0; j < 27; ++j)
        s += direnc[r * 27 + j] * Wd[(256 + j) * 128 + n];
    P[id] = s;
}

// ---------------------------------------------------------------------------
// WMMA GEMM:  C[M,Nc] = act(A[M,K] @ B[K,Nc] + bias)
//   A,B,C f16 row-major (lda=K, ldb=ldc=Nc), f32 accumulation.
//   Block tile 128x64, 8 waves (wave tile 64x16, 4 WMMA accumulators).
//   ROWBIAS: bias indexed [row>>7, n] (per-ray matrix P), else per-column.
// Fragment layouts per CDNA5 ISA 7.12.2:
//   A(16x32 f16): lane L holds M=L%16; halves h: k = 16*(h/8) + 8*(L/16) + h%8
//   B(32x16 f16): lane L holds N=L%16; halves h: k = 16*(h/8) + 8*(L/16) + h%8
//   C(16x16 f32): reg r: m = r + 8*(L/16), n = L%16
// All four A fragments are gathered before the WMMA group so the XDL ops can
// issue back-to-back behind one s_wait_dscnt; next K-slice is prefetched into
// L2 (global_prefetch_b8) while the current tile is staged.
// ---------------------------------------------------------------------------
template<int RELU, int ROWBIAS>
__global__ void __launch_bounds__(256) gemm_wmma_kernel(
    const _Float16* __restrict__ A, const _Float16* __restrict__ B,
    const float* __restrict__ bias, _Float16* __restrict__ C,
    int K, int Nc)
{
    __shared__ __align__(16) _Float16 ldsA[128 * 32];  // A tile, row-major [128][32]
    __shared__ __align__(16) _Float16 ldsB[64 * 32];   // B tile transposed [n][k]

    const int tid   = threadIdx.x;
    const int lane  = tid & 31;
    const int wave  = tid >> 5;
    const int bm    = blockIdx.x * 128;
    const int bn    = blockIdx.y * 64;
    const int wm    = (wave & 1) * 64;
    const int wn    = (wave >> 1) * 16;
    const int lhalf = lane >> 4;   // 0/1
    const int lm    = lane & 15;

    v8f acc[4] = {};

    for (int k0 = 0; k0 < K; k0 += 32) {
        // Speculative L2 prefetch of the next K-slice (global_prefetch_b8).
        if (k0 + 32 < K) {
            __builtin_prefetch(
                &A[(size_t)(bm + (tid >> 1)) * K + (k0 + 32) + (tid & 1) * 16], 0, 1);
            __builtin_prefetch(
                &B[(size_t)(k0 + 32 + (tid >> 3)) * Nc + bn + (tid & 7) * 8], 0, 1);
        }
        // Stage A tile: 128x32 halves; 256 threads x two 16B loads
        #pragma unroll
        for (int i = 0; i < 2; ++i) {
            int idx = tid + i * 256;          // 0..511
            int row = idx >> 2;               // 4 float4 per 32-half row
            int c8  = idx & 3;
            *(float4*)&ldsA[row * 32 + c8 * 8] =
                *(const float4*)&A[(size_t)(bm + row) * K + k0 + c8 * 8];
        }
        // Stage B tile transposed: rows k0..k0+31, cols bn..bn+63
        {
            int krow = tid >> 3;              // 0..31
            int ng   = tid & 7;               // 8-half group along n
            float4 v = *(const float4*)&B[(size_t)(k0 + krow) * Nc + bn + ng * 8];
            const _Float16* hv = (const _Float16*)&v;
            #pragma unroll
            for (int e = 0; e < 8; ++e)
                ldsB[(ng * 8 + e) * 32 + krow] = hv[e];
        }
        __syncthreads();

        FragH fb;
        fb.f4[0] = *(const float4*)&ldsB[(wn + lm) * 32 + lhalf * 8];
        fb.f4[1] = *(const float4*)&ldsB[(wn + lm) * 32 + 16 + lhalf * 8];

        // Gather all A fragments first so the 4 WMMAs issue back-to-back.
        FragH fa[4];
        #pragma unroll
        for (int s = 0; s < 4; ++s) {
            int row = wm + s * 16 + lm;
            fa[s].f4[0] = *(const float4*)&ldsA[row * 32 + lhalf * 8];
            fa[s].f4[1] = *(const float4*)&ldsA[row * 32 + 16 + lhalf * 8];
        }
        #pragma unroll
        for (int s = 0; s < 4; ++s) {
            acc[s] = __builtin_amdgcn_wmma_f32_16x16x32_f16(
                false, fa[s].h, false, fb.h, (short)0, acc[s], false, false);
        }
        __syncthreads();
    }

    // Epilogue: bias (+ ReLU), cast to f16, store.
    #pragma unroll
    for (int s = 0; s < 4; ++s) {
        #pragma unroll
        for (int rr = 0; rr < 8; ++rr) {
            int m = bm + wm + s * 16 + lhalf * 8 + rr;
            int n = bn + wn + lm;
            float v = acc[s][rr];
            if (ROWBIAS) v += bias[(size_t)(m >> 7) * Nc + n];
            else         v += bias[n];
            if (RELU)    v = fmaxf(v, 0.0f);
            C[(size_t)m * Nc + n] = (_Float16)v;
        }
    }
}

// ---------------------------------------------------------------------------
// Sigma (+ optional mus/sigma raw) head: wave per row, 256-wide dot.
// ---------------------------------------------------------------------------
template<int COARSE>
__global__ void __launch_bounds__(256) heads_kernel(
    const _Float16* __restrict__ X, const float* __restrict__ Ws,
    const float* __restrict__ bs, const float* __restrict__ Wm,
    const float* __restrict__ bm, float* __restrict__ rf,
    float* __restrict__ musig, int rowbase)
{
    int row  = blockIdx.x * 8 + (threadIdx.x >> 5);   // local row in chunk
    int lane = threadIdx.x & 31;
    const _Float16* xr = X + (size_t)row * HID + lane * 8;
    float s0 = 0.0f, s1 = 0.0f, s2 = 0.0f;
    #pragma unroll
    for (int e = 0; e < 8; ++e) {
        float xv = (float)xr[e];
        int k = lane * 8 + e;
        s0 += xv * Ws[k];
        if (COARSE) { s1 += xv * Wm[k * 2]; s2 += xv * Wm[k * 2 + 1]; }
    }
    #pragma unroll
    for (int off = 16; off >= 1; off >>= 1) {
        s0 += __shfl_xor(s0, off, 32);
        if (COARSE) { s1 += __shfl_xor(s1, off, 32); s2 += __shfl_xor(s2, off, 32); }
    }
    if (lane == 0) {
        size_t g = (size_t)(rowbase + row);
        rf[g * 4 + 3] = s0 + bs[0];
        if (COARSE) {
            musig[g * 2]     = s1 + bm[0];
            musig[g * 2 + 1] = s2 + bm[1];
        }
    }
}

// ---------------------------------------------------------------------------
// RGB head: wave per row, 128 -> 3.
// ---------------------------------------------------------------------------
__global__ void __launch_bounds__(256) rgbhead_kernel(
    const _Float16* __restrict__ H, const float* __restrict__ Wr,
    const float* __restrict__ br, float* __restrict__ rf, int rowbase)
{
    int row  = blockIdx.x * 8 + (threadIdx.x >> 5);
    int lane = threadIdx.x & 31;
    const _Float16* hr = H + (size_t)row * HDIM + lane * 4;
    float s0 = 0.0f, s1 = 0.0f, s2 = 0.0f;
    #pragma unroll
    for (int e = 0; e < 4; ++e) {
        float hv = (float)hr[e];
        int k = lane * 4 + e;
        s0 += hv * Wr[k * 3];
        s1 += hv * Wr[k * 3 + 1];
        s2 += hv * Wr[k * 3 + 2];
    }
    #pragma unroll
    for (int off = 16; off >= 1; off >>= 1) {
        s0 += __shfl_xor(s0, off, 32);
        s1 += __shfl_xor(s1, off, 32);
        s2 += __shfl_xor(s2, off, 32);
    }
    if (lane == 0) {
        size_t g = (size_t)(rowbase + row);
        rf[g * 4]     = s0 + br[0];
        rf[g * 4 + 1] = s1 + br[1];
        rf[g * 4 + 2] = s2 + br[2];
    }
}

// ---------------------------------------------------------------------------
// Volume rendering: one thread per ray, sequential transmittance scan.
// ---------------------------------------------------------------------------
template<int COARSE>
__global__ void volrender_kernel(
    const float* __restrict__ rf, const float* __restrict__ tfine,
    const float* __restrict__ rd, float* __restrict__ out_rgb,
    float* __restrict__ out_depth, float* __restrict__ out_acc,
    float* __restrict__ weights_out)
{
    int r = blockIdx.x * blockDim.x + threadIdx.x;
    if (r >= NRAYS) return;
    float nrm = sqrtf(rd[r * 3] * rd[r * 3] + rd[r * 3 + 1] * rd[r * 3 + 1]
                    + rd[r * 3 + 2] * rd[r * 3 + 2]);
    float T = 1.0f, R = 0.0f, G = 0.0f, B = 0.0f, depth = 0.0f, acc = 0.0f;
    for (int s = 0; s < NS; ++s) {
        float t0, t1;
        if (COARSE) {
            t0 = NEAR_T + (FAR_T - NEAR_T) * (float)s / (float)NS;
            t1 = NEAR_T + (FAR_T - NEAR_T) * (float)(s + 1) / (float)NS;
        } else {
            t0 = tfine[r * (NS + 1) + s];
            t1 = tfine[r * (NS + 1) + s + 1];
        }
        const float* q = rf + ((size_t)r * NS + s) * 4;
        float sg   = fmaxf(q[3], 0.0f);
        float dist = (t1 - t0) * nrm;
        float a    = 1.0f - expf(-sg * dist);
        float w    = a * T;
        if (COARSE) weights_out[r * NS + s] = w;
        R += w * sigmoidf_(q[0]);
        G += w * sigmoidf_(q[1]);
        B += w * sigmoidf_(q[2]);
        depth += w * 0.5f * (t0 + t1);
        acc   += w;
        T *= (1.0f - a + 1e-10f);
    }
    out_rgb[r * 3] = R; out_rgb[r * 3 + 1] = G; out_rgb[r * 3 + 2] = B;
    if (!COARSE) { out_depth[r] = depth; out_acc[r] = acc; }
}

// ---------------------------------------------------------------------------
// Inverse-CDF fine sampling: streaming searchsorted (u ascending) + erfinv,
// insertion-sort of 129 bin positions. One thread per ray.
// ---------------------------------------------------------------------------
__global__ void samplepdf_kernel(const float* __restrict__ weights,
                                 const float* __restrict__ musig,
                                 float* __restrict__ tfine)
{
    int r = blockIdx.x * blockDim.x + threadIdx.x;
    if (r >= NRAYS) return;
    float pdf[NS];
    float sum = 0.0f;
    for (int i = 0; i < NS; ++i) { pdf[i] = weights[r * NS + i] + 1e-5f; sum += pdf[i]; }
    float invsum = 1.0f / sum;
    for (int i = 0; i < NS; ++i) pdf[i] *= invsum;

    float tf[NS + 1];
    float cdf_i = 0.0f;
    int i = 0;
    for (int j = 0; j <= NS; ++j) {
        float u = (1.0f - 1e-5f) * (float)j / (float)NS;
        while (i < NS - 1 && cdf_i + pdf[i] <= u) { cdf_i += pdf[i]; ++i; }
        float frac = (u - cdf_i) / fmaxf(pdf[i], 1e-10f);
        frac = fminf(fmaxf(frac, 0.0f), 1.0f);
        float m0 = musig[((size_t)r * NS + i) * 2];
        float m1 = musig[((size_t)r * NS + i) * 2 + 1];
        float mu = sigmoidf_(m0);                      // REG_RED = 1
        float sg = (sigmoidf_(m1) + 0.001f) * 2.0f;    // SMOOTH  = 2
        float lt  = acdf_((0.0f - mu) / sg);
        float pib = acdf_((1.0f - mu) / sg) - lt;
        float p = fminf(fmaxf(lt + frac * pib, 1e-5f), 1.0f - 1e-5f);
        float x = mu + sg * 1.4142135623730951f * erfinv_(2.0f * p - 1.0f);
        x = fminf(fmaxf(x, 0.0f), 1.0f);
        float b0 = NEAR_T + (FAR_T - NEAR_T) * (float)i / (float)NS;
        float b1 = NEAR_T + (FAR_T - NEAR_T) * (float)(i + 1) / (float)NS;
        tf[j] = b0 + x * (b1 - b0);
    }
    for (int a = 1; a <= NS; ++a) {          // insertion sort (129 elems)
        float v = tf[a]; int b = a - 1;
        while (b >= 0 && tf[b] > v) { tf[b + 1] = tf[b]; --b; }
        tf[b + 1] = v;
    }
    for (int j = 0; j <= NS; ++j) tfine[r * (NS + 1) + j] = tf[j];
}

// ---------------------------------------------------------------------------
// Host orchestration
// ---------------------------------------------------------------------------
extern "C" void kernel_launch(void* const* d_in, const int* in_sizes, int n_in,
                              void* d_out, int out_size, void* d_ws, size_t ws_size,
                              hipStream_t stream)
{
    (void)in_sizes; (void)n_in; (void)out_size; (void)ws_size;
    const float* ro  = (const float*)d_in[0];
    const float* rd  = (const float*)d_in[1];
    const float* rr  = (const float*)d_in[2];
    const float* cW0 = (const float*)d_in[3];  const float* cb0 = (const float*)d_in[4];
    const float* cWh = (const float*)d_in[5];  const float* cbh = (const float*)d_in[6];
    const float* cWs = (const float*)d_in[7];  const float* cbs = (const float*)d_in[8];
    const float* cWb = (const float*)d_in[9];  const float* cbb = (const float*)d_in[10];
    const float* cWd = (const float*)d_in[11]; const float* cbd = (const float*)d_in[12];
    const float* cWr = (const float*)d_in[13]; const float* cbr = (const float*)d_in[14];
    const float* fW0 = (const float*)d_in[15]; const float* fb0 = (const float*)d_in[16];
    const float* fWh = (const float*)d_in[17]; const float* fbh = (const float*)d_in[18];
    const float* fWs = (const float*)d_in[19]; const float* fbs = (const float*)d_in[20];
    const float* fWb = (const float*)d_in[21]; const float* fbb = (const float*)d_in[22];
    const float* fWd = (const float*)d_in[23]; const float* fbd = (const float*)d_in[24];
    const float* fWr = (const float*)d_in[25]; const float* fbr = (const float*)d_in[26];
    const float* cWm = (const float*)d_in[27]; const float* cbm = (const float*)d_in[28];
    float* out = (float*)d_out;

    // ---- workspace carve (aligned to 256B) ----
    char* wsb = (char*)d_ws;
    size_t off = 0;
    auto carve = [&](size_t bytes) -> char* {
        char* p = wsb + off;
        off += (bytes + 255) & ~(size_t)255;
        return p;
    };
    const int MROWS = NRAYS * NS;
    _Float16* cW0h = (_Float16*)carve((size_t)96 * 256 * 2);
    _Float16* cWhh = (_Float16*)carve((size_t)3 * 256 * 256 * 2);
    _Float16* cWbh = (_Float16*)carve((size_t)256 * 256 * 2);
    _Float16* cWdh = (_Float16*)carve((size_t)256 * 128 * 2);
    _Float16* fW0h = (_Float16*)carve((size_t)96 * 256 * 2);
    _Float16* fWhh = (_Float16*)carve((size_t)3 * 256 * 256 * 2);
    _Float16* fWbh = (_Float16*)carve((size_t)256 * 256 * 2);
    _Float16* fWdh = (_Float16*)carve((size_t)256 * 128 * 2);
    _Float16* enc  = (_Float16*)carve((size_t)MC * ENC_DIM * 2);
    _Float16* xb   = (_Float16*)carve((size_t)MC * HID * 2);
    _Float16* yb   = (_Float16*)carve((size_t)MC * HID * 2);
    float* direnc = (float*)carve((size_t)NRAYS * 27 * 4);
    float* P      = (float*)carve((size_t)CH_RAYS * 128 * 4);
    float* rf     = (float*)carve((size_t)MROWS * 4 * 4);
    float* musig  = (float*)carve((size_t)MROWS * 2 * 4);
    float* wts    = (float*)carve((size_t)NRAYS * NS * 4);
    float* tfine  = (float*)carve((size_t)NRAYS * (NS + 1) * 4);

    // ---- weight conversion ----
    auto cvt = [&](const float* s, _Float16* d, int n) {
        cvt_kernel<<<(n + 255) / 256, 256, 0, stream>>>(s, d, n);
    };
    cvt(cW0, cW0h, 96 * 256);  cvt(cWh, cWhh, 3 * 256 * 256);
    cvt(cWb, cWbh, 256 * 256); cvt(cWd, cWdh, 256 * 128);   // Wd rows 0..255
    cvt(fW0, fW0h, 96 * 256);  cvt(fWh, fWhh, 3 * 256 * 256);
    cvt(fWb, fWbh, 256 * 256); cvt(fWd, fWdh, 256 * 128);

    direnc_kernel<<<(NRAYS + 255) / 256, 256, 0, stream>>>(rd, direnc);

    // ---- one full MLP pass (chunked over rays) ----
    auto mlp_pass = [&](const _Float16* W0h, const float* b0v,
                        const _Float16* Whh, const float* bhv,
                        const _Float16* Wbh, const float* bbv,
                        const _Float16* Wdh, const float* Wdfull, const float* bdv,
                        const float* Wsv, const float* bsv,
                        const float* Wrv, const float* brv,
                        const float* Wmv, const float* bmv,
                        const float* tfine_in, int coarse)
    {
        for (int r0 = 0; r0 < NRAYS; r0 += CH_RAYS) {
            const int rowbase = r0 * NS;
            preproc_kernel<<<MC / 256, 256, 0, stream>>>(ro, rd, rr, tfine_in, enc, rowbase);
            dim3 g0(MC / 128, 256 / 64);
            gemm_wmma_kernel<1, 0><<<g0, 256, 0, stream>>>(enc, W0h, b0v, xb, 96, 256);
            gemm_wmma_kernel<1, 0><<<g0, 256, 0, stream>>>(xb, Whh,                 bhv,       yb, 256, 256);
            gemm_wmma_kernel<1, 0><<<g0, 256, 0, stream>>>(yb, Whh + 1 * 256 * 256, bhv + 256, xb, 256, 256);
            gemm_wmma_kernel<1, 0><<<g0, 256, 0, stream>>>(xb, Whh + 2 * 256 * 256, bhv + 512, yb, 256, 256);
            if (coarse)
                heads_kernel<1><<<MC / 8, 256, 0, stream>>>(yb, Wsv, bsv, Wmv, bmv, rf, musig, rowbase);
            else
                heads_kernel<0><<<MC / 8, 256, 0, stream>>>(yb, Wsv, bsv, nullptr, nullptr, rf, musig, rowbase);
            gemm_wmma_kernel<0, 0><<<g0, 256, 0, stream>>>(yb, Wbh, bbv, xb, 256, 256);  // bottleneck
            pbias_kernel<<<(CH_RAYS * 128) / 256, 256, 0, stream>>>(direnc, Wdfull, bdv, P, r0);
            dim3 g1(MC / 128, 128 / 64);
            gemm_wmma_kernel<1, 1><<<g1, 256, 0, stream>>>(xb, Wdh, P, yb, 256, 128);    // concat layer
            rgbhead_kernel<<<MC / 8, 256, 0, stream>>>(yb, Wrv, brv, rf, rowbase);
        }
    };

    // ---- coarse pass ----
    mlp_pass(cW0h, cb0, cWhh, cbh, cWbh, cbb, cWdh, cWd, cbd,
             cWs, cbs, cWr, cbr, cWm, cbm, nullptr, 1);
    volrender_kernel<1><<<(NRAYS + 255) / 256, 256, 0, stream>>>(
        rf, nullptr, rd, out, nullptr, nullptr, wts);
    samplepdf_kernel<<<(NRAYS + 255) / 256, 256, 0, stream>>>(wts, musig, tfine);

    // ---- fine pass ----
    mlp_pass(fW0h, fb0, fWhh, fbh, fWbh, fbb, fWdh, fWd, fbd,
             fWs, fbs, fWr, fbr, nullptr, nullptr, tfine, 0);
    volrender_kernel<0><<<(NRAYS + 255) / 256, 256, 0, stream>>>(
        rf, tfine, rd, out + NRAYS * 3, out + NRAYS * 6, out + NRAYS * 7, nullptr);
}
